// OptimizedMaterialClassifier_76209899700723
// MI455X (gfx1250) — matrix operand order repaired and checked by reading, not compiled, not run on hardware
//
#include <hip/hip_runtime.h>
#include <hip/hip_bf16.h>

// ---------------- problem constants (from reference) ----------------
#define BGRAPH 512
#define NC 131072
#define EC 1048576
#define NK 32768
#define EK 524288
#define FEAT 1152
#define BN_EPS 1e-5f

typedef __attribute__((ext_vector_type(16))) __bf16 v16bf;
typedef __attribute__((ext_vector_type(8)))  float  v8f;
typedef unsigned short u16;

__device__ __forceinline__ u16 f2bf_u16(float f) {
  // round-to-nearest-even f32 -> bf16
  union { float f; unsigned u; } in; in.f = f;
  unsigned r = in.u + 0x7FFFu + ((in.u >> 16) & 1u);
  return (u16)(r >> 16);
}

// ---------------- operand packing ----------------
// X: f32 [N x K] (row stride lda) -> bf16 row-major [N x K32], zero-padded K
__global__ void pack_x_bf16(const float* __restrict__ X, u16* __restrict__ Xp,
                            int N, int K, int lda, int K32) {
  long i = (long)blockIdx.x * blockDim.x + threadIdx.x;
  if (i >= (long)N * K32) return;
  int n = (int)(i / K32), k = (int)(i % K32);
  float v = (k < K) ? X[(size_t)n * lda + k] : 0.0f;
  Xp[i] = f2bf_u16(v);
}

// W: f32 [K x Fr] -> bf16 fragment-ordered [tn][kb][lane][16], padded to K32 x Fp
// B-fragment layout (32x16 bf16): lane covers col = tn*16+(lane&15),
// k_local = (lane>>4)*8 + (j&7) + (j>>3)*16  (j = 0..15), element j -> VGPR j/2 half j%2.
__global__ void pack_w_bf16(const float* __restrict__ W, u16* __restrict__ Wp,
                            int K, int Fr, int Fp, int K32) {
  long i = (long)blockIdx.x * blockDim.x + threadIdx.x;
  if (i >= (long)Fp * K32) return;
  int j    = (int)(i & 15);
  int lane = (int)((i >> 4) & 31);
  long rest = i >> 9;
  int nkb  = K32 >> 5;
  int kb   = (int)(rest % nkb);
  int tn   = (int)(rest / nkb);
  int k    = kb * 32 + (lane >> 4) * 8 + (j & 7) + (j >> 3) * 16;
  int col  = tn * 16 + (lane & 15);
  float v  = (k < K && col < Fr) ? W[(size_t)k * Fr + col] : 0.0f;
  Wp[i] = f2bf_u16(v);
}

// =====================================================================
// WMMA GEMM, bf16 in / f32 accum. One 16x64 output strip per wave (4 tiles).
// K32 % 32 == 0, Fp % 64 == 0, N % 16 == 0. Branch-free inner loop:
// 10x global_load_b128 (clause) -> 4x v_wmma back-to-back; running byte
// pointers (constant strides) keep address math to one u64 add each.
// epi: (+bias[col]) -> (*scale[col]*scale_mul + shift[col]) -> relu; col < Fr only.
// =====================================================================
__global__ void __launch_bounds__(256)
wmma_gemm_bf16(const u16* __restrict__ Xp, const u16* __restrict__ Wp,
               const float* __restrict__ bias, const float* __restrict__ scale,
               const float* __restrict__ shift, float* __restrict__ Out,
               int N, int K32, int Fp, int Fr, int ldc, int col_off,
               float scale_mul, int do_relu)
{
  const int lane = threadIdx.x & 31;
  const int wave = threadIdx.x >> 5;
  const int tiles_n4 = Fp >> 6;
  const int tiles_m  = N >> 4;
  const int tile = blockIdx.x * 8 + wave;          // wave-uniform -> EXEC stays full
  if (tile >= tiles_m * tiles_n4) return;
  const int tm  = tile / tiles_n4;
  const int tn0 = (tile % tiles_n4) * 4;           // first of 4 col-tiles

  const int half = lane >> 4;
  const int mr   = lane & 15;
  const int nkb  = K32 >> 5;

  union V16 { uint4 q[2]; v16bf v; };

  // running byte pointers (all strides loop-invariant)
  const char* ap = (const char*)(Xp + (size_t)(tm * 16 + mr) * K32) + half * 16;
  const size_t wstride = (size_t)nkb * 1024;       // bytes per col-tile of packed W
  const char* bp0 = (const char*)Wp + (size_t)tn0 * wstride + lane * 32;
  const char* bp1 = bp0 + wstride;
  const char* bp2 = bp1 + wstride;
  const char* bp3 = bp2 + wstride;

  v8f acc[4] = {{}, {}, {}, {}};

  for (int kb = 0; kb < nkb; ++kb) {
    V16 a, b0, b1, b2, b3;
    const uint4* pa = (const uint4*)ap;
    const uint4* p0 = (const uint4*)bp0;
    const uint4* p1 = (const uint4*)bp1;
    const uint4* p2 = (const uint4*)bp2;
    const uint4* p3 = (const uint4*)bp3;
    a.q[0]  = pa[0];  a.q[1]  = pa[2];             // +0B, +32B
    b0.q[0] = p0[0];  b0.q[1] = p0[1];
    b1.q[0] = p1[0];  b1.q[1] = p1[1];
    b2.q[0] = p2[0];  b2.q[1] = p2[1];
    b3.q[0] = p3[0];  b3.q[1] = p3[1];
    __builtin_prefetch(ap + 64, 0, 3);             // global_prefetch next A block
    acc[0] = __builtin_amdgcn_wmma_f32_16x16x32_bf16(false, a.v, false, b0.v,
                                                     (short)0, acc[0], false, false);
    acc[1] = __builtin_amdgcn_wmma_f32_16x16x32_bf16(false, a.v, false, b1.v,
                                                     (short)0, acc[1], false, false);
    acc[2] = __builtin_amdgcn_wmma_f32_16x16x32_bf16(false, a.v, false, b2.v,
                                                     (short)0, acc[2], false, false);
    acc[3] = __builtin_amdgcn_wmma_f32_16x16x32_bf16(false, a.v, false, b3.v,
                                                     (short)0, acc[3], false, false);
    ap  += 64;
    bp0 += 1024; bp1 += 1024; bp2 += 1024; bp3 += 1024;
  }

#pragma unroll
  for (int j = 0; j < 4; ++j) {
    int col = (tn0 + j) * 16 + mr;
    if (col < Fr) {
      float bv = bias  ? bias[col]              : 0.0f;
      float sc = scale ? scale[col] * scale_mul : 1.0f;
      float sh = shift ? shift[col]             : 0.0f;
#pragma unroll
      for (int v = 0; v < 8; ++v) {
        int r = tm * 16 + v + 8 * half;            // C/D layout: M = v + 8*half
        float val = acc[j][v] + bv;
        if (scale)   val = val * sc + sh;
        if (do_relu) val = fmaxf(val, 0.0f);
        Out[(size_t)r * ldc + col_off + col] = val;
      }
    }
  }
}

// ---------------- degree / rsqrt ----------------
__global__ void deg_kernel(const int* __restrict__ dst, float* __restrict__ deg, int E) {
  int i = blockIdx.x * blockDim.x + threadIdx.x;
  if (i < E) atomicAdd(&deg[dst[i]], 1.0f);
}
__global__ void dinv_kernel(float* __restrict__ deg, int n) {
  int i = blockIdx.x * blockDim.x + threadIdx.x;
  if (i < n) deg[i] = __frsqrt_rn(deg[i] + 1.0f);
}

// ---------------- edge gather + atomic scatter-add ----------------
__global__ void edge_agg(const float* __restrict__ h, const int* __restrict__ src,
                         const int* __restrict__ dst, const float* __restrict__ dinv,
                         float* __restrict__ agg, int E, int F) {
  long tid = (long)blockIdx.x * blockDim.x + threadIdx.x;
  int fchunks = F >> 2;
  if (tid >= (long)E * fchunks) return;
  int e  = (int)(tid / fchunks);
  int f4 = (int)(tid % fchunks) << 2;
  int s = src[e], d = dst[e];
  float nrm = dinv[s] * dinv[d];
  const float4 v = *(const float4*)(h + (size_t)s * F + f4);
  float* o = agg + (size_t)d * F + f4;
  atomicAdd(o + 0, v.x * nrm);
  atomicAdd(o + 1, v.y * nrm);
  atomicAdd(o + 2, v.z * nrm);
  atomicAdd(o + 3, v.w * nrm);
}

// ---------------- GCN epilogue: out = relu(agg + h*dinv^2 + b [+resid]) ----------------
__global__ void gcn_finish(const float* __restrict__ agg, const float* __restrict__ h,
                           const float* __restrict__ dinv, const float* __restrict__ bias,
                           const float* __restrict__ resid, float* __restrict__ out,
                           int n, int F) {
  long tid = (long)blockIdx.x * blockDim.x + threadIdx.x;
  if (tid >= (long)n * F) return;
  int i = (int)(tid / F), f = (int)(tid % F);
  float di = dinv[i];
  float v = agg[tid] + h[tid] * di * di + bias[f];
  if (resid) v += resid[tid];
  out[tid] = fmaxf(v, 0.0f);
}

// ---------------- mean/max pool over contiguous graphs ----------------
__global__ void pool_kernel(const float* __restrict__ x, float* __restrict__ feats,
                            int npg, int F, int ldc, int col_off) {
  int b = blockIdx.x;
  const float* base = x + (size_t)b * npg * F;
  float inv = 1.0f / (float)npg;
  for (int f = threadIdx.x; f < F; f += blockDim.x) {
    float s = 0.0f, mx = -3.4e38f;
    for (int i = 0; i < npg; ++i) {
      float v = base[(size_t)i * F + f];
      s += v;
      mx = fmaxf(mx, v);
    }
    feats[(size_t)b * ldc + col_off + f]     = s * inv;
    feats[(size_t)b * ldc + col_off + F + f] = mx;
  }
}

__global__ void add_inplace(float* __restrict__ a, const float* __restrict__ b, int n) {
  int i = blockIdx.x * blockDim.x + threadIdx.x;
  if (i < n) a[i] += b[i];
}

// extract [B,2] logits out of padded [B,64] buffer
__global__ void extract_logits(const float* __restrict__ pad, float* __restrict__ out, int n) {
  int i = blockIdx.x * blockDim.x + threadIdx.x;
  if (i < n) out[i] = pad[(size_t)(i >> 1) * 64 + (i & 1)];
}

// =====================================================================
// host-side orchestration
// =====================================================================
static inline void gemm(hipStream_t st, const float* X, int lda, const float* W,
                        int K, int Fr, int Fp,
                        const float* bias, const float* scale, const float* shift,
                        float* Out, int N, int ldc, int col_off, float smul, int relu,
                        u16* Xp, u16* Wp) {
  int K32 = (K + 31) & ~31;
  long xe = (long)N * K32;
  long we = (long)Fp * K32;
  pack_x_bf16<<<(int)((xe + 255) / 256), 256, 0, st>>>(X, Xp, N, K, lda, K32);
  pack_w_bf16<<<(int)((we + 255) / 256), 256, 0, st>>>(W, Wp, K, Fr, Fp, K32);
  int tiles = (N >> 4) * (Fp >> 6);
  wmma_gemm_bf16<<<(tiles + 7) / 8, 256, 0, st>>>(Xp, Wp, bias, scale, shift, Out,
                                                  N, K32, Fp, Fr, ldc, col_off, smul, relu);
}

static inline void run_edge_agg(hipStream_t s, const float* h, const int* src, const int* dst,
                                const float* dinv, float* agg, int E, int F) {
  long total = (long)E * (F >> 2);
  edge_agg<<<(int)((total + 255) / 256), 256, 0, s>>>(h, src, dst, dinv, agg, E, F);
}

static inline void run_finish(hipStream_t s, float* agg_out, const float* h, const float* dinv,
                              const float* bias, const float* resid, int n, int F) {
  long total = (long)n * F;
  gcn_finish<<<(int)((total + 255) / 256), 256, 0, s>>>(agg_out, h, dinv, bias, resid,
                                                        agg_out, n, F);
}

extern "C" void kernel_launch(void* const* d_in, const int* in_sizes, int n_in,
                              void* d_out, int out_size, void* d_ws, size_t ws_size,
                              hipStream_t stream) {
  // -------- inputs (setup_inputs dict order; params dict flattened in order) --------
  const float* crystal_x = (const float*)d_in[0];                     // [NC,3]
  const int*   c_ei      = (const int*)d_in[1];                       // [2,EC]
  const float* kspace_x  = (const float*)d_in[3];                     // [NK,64]
  const int*   k_ei      = (const int*)d_in[4];                       // [2,EK]
  const float* scal_f    = (const float*)d_in[6];                     // [B,10]
  const float* deco_f    = (const float*)d_in[7];                     // [B,64]
  const int* c_src = c_ei, *c_dst = c_ei + EC;
  const int* k_src = k_ei, *k_dst = k_ei + EK;

  int p = 8;
  const float *cW1=(const float*)d_in[p++], *cb1=(const float*)d_in[p++];
  const float *cW2=(const float*)d_in[p++], *cb2=(const float*)d_in[p++];
  const float *cW3=(const float*)d_in[p++], *cb3=(const float*)d_in[p++];
  const float *kW1=(const float*)d_in[p++], *kb1=(const float*)d_in[p++];
  const float *kW2=(const float*)d_in[p++], *kb2=(const float*)d_in[p++];
  const float *kW3=(const float*)d_in[p++], *kb3=(const float*)d_in[p++];
  const float *sW1=(const float*)d_in[p++], *sb1=(const float*)d_in[p++];
  const float *sg =(const float*)d_in[p++], *sbe=(const float*)d_in[p++];
  const float *sW2=(const float*)d_in[p++], *sb2=(const float*)d_in[p++];
  const float *dW1=(const float*)d_in[p++], *db1=(const float*)d_in[p++];
  const float *dg =(const float*)d_in[p++], *dbe=(const float*)d_in[p++];
  const float *dW2=(const float*)d_in[p++], *db2=(const float*)d_in[p++];
  const float *Wv =(const float*)d_in[p++], *bv =(const float*)d_in[p++];
  const float *Wo =(const float*)d_in[p++], *bo =(const float*)d_in[p++];
  const float *clW1=(const float*)d_in[p++], *clb1=(const float*)d_in[p++];
  const float *clg1=(const float*)d_in[p++], *clbe1=(const float*)d_in[p++];
  const float *clW2=(const float*)d_in[p++], *clb2=(const float*)d_in[p++];
  const float *clg2=(const float*)d_in[p++], *clbe2=(const float*)d_in[p++];
  const float *clW3=(const float*)d_in[p++], *clb3=(const float*)d_in[p++];

  float* out = (float*)d_out;                                         // [B,2] logits

  // -------- workspace layout --------
  const size_t R = 16777216;                       // 64 MiB region (>= NC*128, NK*256)
  float* ws    = (float*)d_ws;
  float* H     = ws;                               // gemm output h
  float* XA    = ws + R;                           // ping
  float* XB    = ws + 2 * R;                       // pong
  float* dinv  = ws + 3 * R;                       // [max(NC,NK)]
  float* feats = dinv  + NC;                       // [B,1152]
  float* tmpV  = feats + (size_t)BGRAPH * FEAT;    // [B,1152]
  float* tmpA  = tmpV  + (size_t)BGRAPH * FEAT;    // [B,1152]
  float* s_h   = tmpA  + (size_t)BGRAPH * FEAT;    // [B,64]
  float* d_h   = s_h   + (size_t)BGRAPH * 64;      // [B,128]
  float* c1    = d_h   + (size_t)BGRAPH * 128;     // [B,256]
  float* c2    = c1    + (size_t)BGRAPH * 256;     // [B,128]
  float* lpad  = c2    + (size_t)BGRAPH * 128;     // [B,64] padded logits
  u16*   Xp    = (u16*)(lpad + (size_t)BGRAPH * 64);  // bf16 packed X (<= N*K32 = 16.8M)
  u16*   Wp    = Xp + (size_t)16777216;               // bf16 packed W (<= 1152*1152)

  const float bnmul = 1.0f / sqrtf(1.0f + BN_EPS); // folded eval-mode BN scale

  // ============== crystal GCN stack ==============
  hipMemsetAsync(dinv, 0, NC * sizeof(float), stream);
  deg_kernel<<<(EC + 255) / 256, 256, 0, stream>>>(c_dst, dinv, EC);
  dinv_kernel<<<(NC + 255) / 256, 256, 0, stream>>>(dinv, NC);

  // layer1: 3 -> 64
  gemm(stream, crystal_x, 3, cW1, 3, 64, 64, nullptr, nullptr, nullptr, H, NC, 64, 0, 1.f, 0, Xp, Wp);
  hipMemsetAsync(XA, 0, (size_t)NC * 64 * sizeof(float), stream);
  run_edge_agg(stream, H, c_src, c_dst, dinv, XA, EC, 64);
  run_finish(stream, XA, H, dinv, cb1, nullptr, NC, 64);
  // layer2: 64 -> 128
  gemm(stream, XA, 64, cW2, 64, 128, 128, nullptr, nullptr, nullptr, H, NC, 128, 0, 1.f, 0, Xp, Wp);
  hipMemsetAsync(XB, 0, (size_t)NC * 128 * sizeof(float), stream);
  run_edge_agg(stream, H, c_src, c_dst, dinv, XB, EC, 128);
  run_finish(stream, XB, H, dinv, cb2, nullptr, NC, 128);
  // layer3: 128 -> 128 (+residual XB)
  gemm(stream, XB, 128, cW3, 128, 128, 128, nullptr, nullptr, nullptr, H, NC, 128, 0, 1.f, 0, Xp, Wp);
  hipMemsetAsync(XA, 0, (size_t)NC * 128 * sizeof(float), stream);
  run_edge_agg(stream, H, c_src, c_dst, dinv, XA, EC, 128);
  run_finish(stream, XA, H, dinv, cb3, XB, NC, 128);
  // pool -> feats cols [0,256)
  pool_kernel<<<BGRAPH, 128, 0, stream>>>(XA, feats, NC / BGRAPH, 128, FEAT, 0);

  // ============== kspace GCN stack ==============
  hipMemsetAsync(dinv, 0, NK * sizeof(float), stream);
  deg_kernel<<<(EK + 255) / 256, 256, 0, stream>>>(k_dst, dinv, EK);
  dinv_kernel<<<(NK + 255) / 256, 256, 0, stream>>>(dinv, NK);

  // layer1: 64 -> 128
  gemm(stream, kspace_x, 64, kW1, 64, 128, 128, nullptr, nullptr, nullptr, H, NK, 128, 0, 1.f, 0, Xp, Wp);
  hipMemsetAsync(XA, 0, (size_t)NK * 128 * sizeof(float), stream);
  run_edge_agg(stream, H, k_src, k_dst, dinv, XA, EK, 128);
  run_finish(stream, XA, H, dinv, kb1, nullptr, NK, 128);
  // layer2: 128 -> 256
  gemm(stream, XA, 128, kW2, 128, 256, 256, nullptr, nullptr, nullptr, H, NK, 256, 0, 1.f, 0, Xp, Wp);
  hipMemsetAsync(XB, 0, (size_t)NK * 256 * sizeof(float), stream);
  run_edge_agg(stream, H, k_src, k_dst, dinv, XB, EK, 256);
  run_finish(stream, XB, H, dinv, kb2, nullptr, NK, 256);
  // layer3: 256 -> 256 (+residual XB)
  gemm(stream, XB, 256, kW3, 256, 256, 256, nullptr, nullptr, nullptr, H, NK, 256, 0, 1.f, 0, Xp, Wp);
  hipMemsetAsync(XA, 0, (size_t)NK * 256 * sizeof(float), stream);
  run_edge_agg(stream, H, k_src, k_dst, dinv, XA, EK, 256);
  run_finish(stream, XA, H, dinv, kb3, XB, NK, 256);
  // pool -> feats cols [256,768)
  pool_kernel<<<BGRAPH, 256, 0, stream>>>(XA, feats, NK / BGRAPH, 256, FEAT, 256);

  // ============== scalar / decomposition encoders ==============
  gemm(stream, scal_f, 10, sW1, 10, 64, 64, sb1, sg, sbe, s_h, BGRAPH, 64, 0, bnmul, 1, Xp, Wp);
  gemm(stream, s_h, 64, sW2, 64, 128, 128, sb2, nullptr, nullptr, feats, BGRAPH, FEAT, 768, 1.f, 0, Xp, Wp);
  gemm(stream, deco_f, 64, dW1, 64, 128, 128, db1, dg, dbe, d_h, BGRAPH, 128, 0, bnmul, 1, Xp, Wp);
  gemm(stream, d_h, 128, dW2, 128, 256, 256, db2, nullptr, nullptr, feats, BGRAPH, FEAT, 896, 1.f, 0, Xp, Wp);

  // ============== seq-len-1 attention: feats += (feats@Wv+bv)@Wo+bo ==============
  gemm(stream, feats, FEAT, Wv, FEAT, FEAT, FEAT, bv, nullptr, nullptr, tmpV, BGRAPH, FEAT, 0, 1.f, 0, Xp, Wp);
  gemm(stream, tmpV, FEAT, Wo, FEAT, FEAT, FEAT, bo, nullptr, nullptr, tmpA, BGRAPH, FEAT, 0, 1.f, 0, Xp, Wp);
  add_inplace<<<(BGRAPH * FEAT + 255) / 256, 256, 0, stream>>>(feats, tmpA, BGRAPH * FEAT);

  // ============== classifier head ==============
  gemm(stream, feats, FEAT, clW1, FEAT, 256, 256, clb1, clg1, clbe1, c1, BGRAPH, 256, 0, bnmul, 1, Xp, Wp);
  gemm(stream, c1, 256, clW2, 256, 128, 128, clb2, clg2, clbe2, c2, BGRAPH, 128, 0, bnmul, 1, Xp, Wp);
  gemm(stream, c2, 128, clW3, 128, 2, 64, clb3, nullptr, nullptr, lpad, BGRAPH, 64, 0, 1.f, 0, Xp, Wp);
  extract_logits<<<(BGRAPH * 2 + 255) / 256, 256, 0, stream>>>(lpad, out, BGRAPH * 2);

  (void)in_sizes; (void)n_in; (void)out_size; (void)ws_size;
}
